// MoEProjection_20590073217580
// MI455X (gfx1250) — matrix operand order repaired and checked by reading, not compiled, not run on hardware
//
#include <hip/hip_runtime.h>
#include <hip/hip_bf16.h>
#include <math.h>
#include <stdint.h>

#define DEVI __device__ __forceinline__

typedef __attribute__((ext_vector_type(16))) _Float16 v16h;
typedef __attribute__((ext_vector_type(8)))  _Float16 v8h;
typedef __attribute__((ext_vector_type(8)))  float    v8f;
typedef __attribute__((ext_vector_type(4)))  unsigned int u32x4;
typedef __attribute__((ext_vector_type(8)))  int      i32x8;
typedef __attribute__((ext_vector_type(4)))  int      i32x4;

#if defined(__has_builtin)
#if __has_builtin(__builtin_amdgcn_tensor_load_to_lds)
#define HAVE_TDM 1
#endif
#if __has_builtin(__builtin_amdgcn_s_wait_tensorcnt)
#define HAVE_TCNT 1
#endif
#endif

namespace moecfg {
constexpr int E = 4, B = 8, L = 768, C = 1024, H = 768, Q = 144, NH = 12, O = 4096, NL = 2;
constexpr int MLP = 4 * H;      // 3072
constexpr int HD = H / NH;      // 64
constexpr int KV = 912;         // sum over stages of (Sq + Si)
constexpr int BQ = B * Q;       // 1152
constexpr int BL = B * L;       // 6144
}
using namespace moecfg;

// ---------------------------------------------------------------------------
// Wave helpers (wave32)
// ---------------------------------------------------------------------------
DEVI float wred_sum(float v) {
#pragma unroll
  for (int m = 16; m >= 1; m >>= 1) v += __shfl_xor(v, m, 32);
  return v;
}
DEVI float hred_sum(float v) {  // reduce within 16-lane half
#pragma unroll
  for (int m = 8; m >= 1; m >>= 1) v += __shfl_xor(v, m, 32);
  return v;
}
DEVI float hred_max(float v) {
#pragma unroll
  for (int m = 8; m >= 1; m >>= 1) v = fmaxf(v, __shfl_xor(v, m, 32));
  return v;
}

// ---------------------------------------------------------------------------
// WMMA fragment loaders (CDNA5 16x16x32 f16, wave32, this code's contiguous k order for A and B)
// A 16x32: lanes 0-15: M=lane, K=0..15; lanes 16-31: M=lane-16, K=16..31
// ---------------------------------------------------------------------------
DEVI v16h load_a_frag(const _Float16* __restrict__ Arow, int lda) {
  int l = threadIdx.x & 31;
  const _Float16* p = Arow + (l & 15) * lda + ((l >> 4) << 4);
  v8h lo = *(const v8h*)p;
  v8h hi = *(const v8h*)(p + 8);
  v16h r;
#pragma unroll
  for (int i = 0; i < 8; ++i) { r[i] = lo[i]; r[i + 8] = hi[i]; }
  return r;
}
// B 32x16 from a TRANSPOSED weight Wt[N][K]: lane n=l&15 holds 16 contiguous K.
DEVI v16h load_b_frag(const _Float16* __restrict__ Bt, int ldb) {
  int l = threadIdx.x & 31;
  return *(const v16h*)(Bt + (long)(l & 15) * ldb + ((l >> 4) << 4));
}
// B 32x16 gathered from row-major V[token][dim] (stride ldv between K rows).
DEVI v16h load_b_frag_strided(const _Float16* __restrict__ V, int ldv,
                              int row0, int maxrow, int col) {
  int l = threadIdx.x & 31;
  int n = l & 15, kb = (l >> 4) << 4;
  v16h r;
#pragma unroll
  for (int j = 0; j < 16; ++j) {
    int rr = row0 + kb + j;
    rr = rr > maxrow ? maxrow : rr;
    r[j] = V[(long)rr * ldv + col + n];
  }
  return r;
}
DEVI v8f wmma16(v16h a, v16h b, v8f c) {
  return __builtin_amdgcn_wmma_f32_16x16x32_f16(false, a, false, b, (short)0, c,
                                                false, false);
}

// ---------------------------------------------------------------------------
// Tensor Data Mover staging of the block's shared B tile (64 N-rows x 32 K).
// One TDM per k-step (issued by wave 0), double-buffered in LDS, TENSORcnt.
// ---------------------------------------------------------------------------
DEVI unsigned lds_off_of(const void* p) {
  return (unsigned)(size_t)(__attribute__((address_space(3))) const void*)p;
}

#if defined(HAVE_TDM)
DEVI void tdm_stage_b(const _Float16* gsrc, unsigned lds_off, int ldK) {
  unsigned long long ga = (unsigned long long)(size_t)gsrc;
  u32x4 g0;
  g0[0] = 1u;                                   // count=1, user descriptor
  g0[1] = lds_off;                              // LDS byte address of tile
  g0[2] = (unsigned)ga;                         // global_addr[31:0]
  g0[3] = (unsigned)((ga >> 32) & 0x01ffffffu) | (2u << 30);  // [56:32] | type=2
  i32x8 g1;
  g1[0] = (int)(1u << 16);                      // data_size = 1 -> 2-byte elems
  g1[1] = (int)(32u << 16);                     // tensor_dim0 = 32
  g1[2] = (int)(64u << 16);                     // tensor_dim1 = 64
  g1[3] = (int)(32u << 16);                     // tile_dim0 = 32
  g1[4] = 64;                                   // tile_dim1 = 64 (tile_dim2 = 0)
  g1[5] = ldK;                                  // tensor_dim0_stride[31:0] = K
  g1[6] = 0;                                    // stride hi / dim1_stride lo
  g1[7] = 0;
  i32x4 z4 = {0, 0, 0, 0};
  i32x8 z8 = {0, 0, 0, 0, 0, 0, 0, 0};
  // 6-arg form (this toolchain): (g0, g1, g2, g3, g_extra, cpol)
  __builtin_amdgcn_tensor_load_to_lds(g0, g1, z4, z4, z8, 0);
}
#endif

DEVI void stage_b(const _Float16* gsrc, _Float16* ldst, int ldK, int wave, int tid) {
#if defined(HAVE_TDM)
  if (wave == 0) tdm_stage_b(gsrc, lds_off_of(ldst), ldK);
  (void)tid;
#else
  int row = tid >> 1, c0 = (tid & 1) * 16;
  *(v8h*)(ldst + row * 32 + c0) = *(const v8h*)(gsrc + (long)row * ldK + c0);
  *(v8h*)(ldst + row * 32 + c0 + 8) = *(const v8h*)(gsrc + (long)row * ldK + c0 + 8);
#endif
}

DEVI void stage_commit(int wave) {
#if defined(HAVE_TDM)
  if (wave == 0) {
#if defined(HAVE_TCNT)
    __builtin_amdgcn_s_wait_tensorcnt((short)0);
#else
    asm volatile("s_wait_tensorcnt 0x0" ::: "memory");
#endif
  }
#endif
  __syncthreads();
}

// ---------------------------------------------------------------------------
// Weight transpose + f32->f16 : src[batch][R][Cn] -> dst[batch][Cn][R]
// ---------------------------------------------------------------------------
__global__ void k_transpose16(const float* __restrict__ src, _Float16* __restrict__ dst,
                              int R, int Cn, long nb) {
  long per = (long)R * Cn, total = per * nb;
  for (long i = (long)blockIdx.x * blockDim.x + threadIdx.x; i < total;
       i += (long)gridDim.x * blockDim.x) {
    long bat = i / per, rem = i % per;
    long r = rem / Cn, c = rem % Cn;
    dst[bat * per + c * R + r] = (_Float16)src[i];
  }
}

__global__ void k_f32_to_f16(const float* __restrict__ s, _Float16* __restrict__ d, long n) {
  for (long i = (long)blockIdx.x * blockDim.x + threadIdx.x; i < n;
       i += (long)gridDim.x * blockDim.x)
    d[i] = (_Float16)s[i];
}

__global__ void k_bcast_query(const float* __restrict__ q, float* __restrict__ t, long total) {
  for (long i = (long)blockIdx.x * blockDim.x + threadIdx.x; i < total;
       i += (long)gridDim.x * blockDim.x) {
    long qh = i % ((long)Q * H);
    long e = i / ((long)B * Q * H);
    t[i] = q[e * (long)Q * H + qh];
  }
}

// ---------------------------------------------------------------------------
// Gating: per token LN(C=1024) -> logits[E] -> clip/softmax -> gated f16 [E,B,L,C]
// ---------------------------------------------------------------------------
__global__ __launch_bounds__(256) void k_gate(
    const float* __restrict__ x, const float* __restrict__ glw,
    const float* __restrict__ glb, const float* __restrict__ gw,
    const float* __restrict__ gb, _Float16* __restrict__ gated) {
  int l = threadIdx.x & 31;
  int row = blockIdx.x * 8 + (threadIdx.x >> 5);  // token in [0, B*L)
  const float* xr = x + (long)row * C;
  float xv[32], s = 0.f, s2 = 0.f;
#pragma unroll
  for (int j = 0; j < 32; ++j) {
    float v = xr[l + 32 * j];
    xv[j] = v; s += v; s2 += v * v;
  }
  s = wred_sum(s); s2 = wred_sum(s2);
  float mean = s * (1.0f / C);
  float var = s2 * (1.0f / C) - mean * mean;
  float rs = rsqrtf(var + 1e-5f);
  float lg[4] = {0.f, 0.f, 0.f, 0.f};
#pragma unroll
  for (int j = 0; j < 32; ++j) {
    int c = l + 32 * j;
    float ln = (xv[j] - mean) * rs * glw[c] + glb[c];
#pragma unroll
    for (int e = 0; e < 4; ++e) lg[e] += ln * gw[c * 4 + e];
  }
#pragma unroll
  for (int e = 0; e < 4; ++e) {
    lg[e] = wred_sum(lg[e]) + gb[e];
    lg[e] = fminf(fmaxf(lg[e], -15.f), 15.f);
  }
  float mx = fmaxf(fmaxf(lg[0], lg[1]), fmaxf(lg[2], lg[3]));
  float we[4], ssum = 0.f;
#pragma unroll
  for (int e = 0; e < 4; ++e) { we[e] = __expf(lg[e] - mx); ssum += we[e]; }
  float inv = 1.f / ssum;
#pragma unroll
  for (int e = 0; e < 4; ++e) we[e] *= inv;
#pragma unroll
  for (int j = 0; j < 32; ++j) {
    int c = l + 32 * j;
#pragma unroll
    for (int e = 0; e < 4; ++e)
      gated[((long)e * BL + row) * C + c] = (_Float16)(xv[j] * we[e]);
  }
}

// ---------------------------------------------------------------------------
// Generic LN over H=768, f32 src -> f16 dst; per-(expert,layer) params.
// ---------------------------------------------------------------------------
__global__ __launch_bounds__(256) void k_ln_rows(
    const float* __restrict__ src, _Float16* __restrict__ dst,
    const float* __restrict__ w, const float* __restrict__ b,
    int rowsPerExpert, int li) {
  int l = threadIdx.x & 31;
  int row = blockIdx.x * 8 + (threadIdx.x >> 5);
  int e = row / rowsPerExpert;
  const float* pr = src + (long)row * H;
  const float* pw = w + (long)(e * NL + li) * H;
  const float* pb = b + (long)(e * NL + li) * H;
  float xv[24], s = 0.f, s2 = 0.f;
#pragma unroll
  for (int j = 0; j < 24; ++j) {
    float v = pr[l + 32 * j];
    xv[j] = v; s += v; s2 += v * v;
  }
  s = wred_sum(s); s2 = wred_sum(s2);
  float mean = s * (1.0f / H);
  float rs = rsqrtf(s2 * (1.0f / H) - mean * mean + 1e-5f);
#pragma unroll
  for (int j = 0; j < 24; ++j) {
    int c = l + 32 * j;
    dst[(long)row * H + c] = (_Float16)((xv[j] - mean) * rs * pw[c] + pb[c]);
  }
}

// kv rows gather from broadcast query (f32) or h16 (f16), then LN -> kvn16.
__global__ __launch_bounds__(256) void k_ln_kv(
    const _Float16* __restrict__ h16, const float* __restrict__ query,
    const float* __restrict__ w, const float* __restrict__ b,
    _Float16* __restrict__ kvn, int li) {
  int l = threadIdx.x & 31;
  int row = blockIdx.x * 8 + (threadIdx.x >> 5);  // [0, E*B*KV)
  int e = row / (B * KV);
  int rem = row % (B * KV);
  int bb = rem / KV, tok = rem % KV;
  const int sbase[3] = {0, 320, 624}, sq[3] = {64, 48, 32},
            qo[3] = {0, 64, 112}, lo[3] = {0, 256, 512};
  int stage = tok >= 624 ? 2 : (tok >= 320 ? 1 : 0);
  int u = tok - sbase[stage];
  float xv[24], s = 0.f, s2 = 0.f;
  if (u < sq[stage]) {
    const float* p = query + ((long)e * Q + qo[stage] + u) * H;
#pragma unroll
    for (int j = 0; j < 24; ++j) {
      float v = p[l + 32 * j];
      xv[j] = v; s += v; s2 += v * v;
    }
  } else {
    const _Float16* p =
        h16 + (((long)e * B + bb) * L + lo[stage] + (u - sq[stage])) * H;
#pragma unroll
    for (int j = 0; j < 24; ++j) {
      float v = (float)p[l + 32 * j];
      xv[j] = v; s += v; s2 += v * v;
    }
  }
  s = wred_sum(s); s2 = wred_sum(s2);
  float mean = s * (1.0f / H);
  float rs = rsqrtf(s2 * (1.0f / H) - mean * mean + 1e-5f);
  const float* pw = w + (long)(e * NL + li) * H;
  const float* pb = b + (long)(e * NL + li) * H;
#pragma unroll
  for (int j = 0; j < 24; ++j) {
    int c = l + 32 * j;
    kvn[(long)row * H + c] = (_Float16)((xv[j] - mean) * rs * pw[c] + pb[c]);
  }
}

// ---------------------------------------------------------------------------
// Generic WMMA GEMM: D = A[M,K] * Wt[N,K]^T + bias, epilogue variants.
// Block = 128 (4 waves), block tile 128(M) x 64(N); wave tile 32x64.
// B tile (shared by all 4 waves) staged via TDM into double-buffered LDS.
// A fragments double-buffered in registers for load/WMMA overlap. grid.z = e.
// ---------------------------------------------------------------------------
enum { EP_F16 = 0, EP_F16_GELU = 1, EP_RESID = 2 };

template <int MODE>
__global__ __launch_bounds__(128) void k_gemm(
    const _Float16* __restrict__ A, const _Float16* __restrict__ Bt,
    const float* __restrict__ bias, _Float16* __restrict__ O16,
    float* __restrict__ O32, const float* __restrict__ scale,
    int M, int N, int K, long sA, long sB, long sBias, long sOut, long sScale) {
  __shared__ __align__(64) _Float16 bsh[2][64 * 32];
  int e = blockIdx.z;
  int tid = threadIdx.x, wave = tid >> 5, l = tid & 31;
  int m0 = blockIdx.y * 128 + wave * 32;
  int n0 = blockIdx.x * 64;
  const _Float16* Ar = A + e * sA + (long)m0 * K;
  const _Float16* Bb = Bt + e * sB + (long)n0 * K;  // rows n0..n0+63 of Wt
  stage_b(Bb, &bsh[0][0], K, wave, tid);            // k-step 0
  v8f acc[2][4] = {};
  v16h a0 = load_a_frag(Ar, K);
  v16h a1 = load_a_frag(Ar + (long)16 * K, K);
  stage_commit(wave);                               // bsh[0] ready
  int nsteps = K >> 5;
  for (int s = 0; s < nsteps; ++s) {
    int buf = s & 1;
    if (s + 1 < nsteps)
      stage_b(Bb + (s + 1) * 32, &bsh[buf ^ 1][0], K, wave, tid);
    v16h na0 = a0, na1 = a1;
    if (s + 1 < nsteps) {
      na0 = load_a_frag(Ar + (s + 1) * 32, K);
      na1 = load_a_frag(Ar + (long)16 * K + (s + 1) * 32, K);
    }
    v16h bf[4];
#pragma unroll
    for (int t = 0; t < 4; ++t) bf[t] = load_b_frag(&bsh[buf][t * 16 * 32], 32);
#pragma unroll
    for (int t = 0; t < 4; ++t) {
      acc[0][t] = wmma16(a0, bf[t], acc[0][t]);
      acc[1][t] = wmma16(a1, bf[t], acc[1][t]);
    }
    stage_commit(wave);  // next buffer's TDM done + release buf for rewrite
    a0 = na0; a1 = na1;
  }
  const float* bi = bias + e * sBias;
  int nl = l & 15, mb = (l >> 4) * 8;
#pragma unroll
  for (int i = 0; i < 2; ++i) {
#pragma unroll
    for (int t = 0; t < 4; ++t) {
      int n = n0 + 16 * t + nl;
      float bv = bi[n];
      float sc = (MODE == EP_RESID) ? scale[e * sScale + n] : 0.f;
#pragma unroll
      for (int r = 0; r < 8; ++r) {
        long m = m0 + 16 * i + mb + r;
        float v = acc[i][t][r] + bv;
        if (MODE == EP_F16) {
          O16[e * sOut + m * N + n] = (_Float16)v;
        } else if (MODE == EP_F16_GELU) {
          float g = 0.5f * v * (1.0f + erff(v * 0.70710678118f));
          O16[e * sOut + m * N + n] = (_Float16)g;
        } else {  // EP_RESID
          long idx = e * sOut + m * N + n;
          O32[idx] = O32[idx] + sc * v;
        }
      }
    }
  }
}

// Output GEMM summed over experts: mixed[BQ,O] = sum_e tok16[e] @ out_wt[e]^T + sum_e bias
__global__ __launch_bounds__(128) void k_gemm_out(
    const _Float16* __restrict__ tok, const _Float16* __restrict__ Wt,
    const float* __restrict__ outb, float* __restrict__ mixed) {
  int wave = threadIdx.x >> 5, l = threadIdx.x & 31;
  int m0 = blockIdx.y * 64 + wave * 16;
  int n0 = blockIdx.x * 64;
  v8f acc[4] = {};
  for (int e = 0; e < E; ++e) {
    const _Float16* Ar = tok + ((long)e * BQ + m0) * H;
    const _Float16* Bb = Wt + (long)e * O * H;
    for (int k = 0; k < H; k += 32) {
      if (k + 64 < H) __builtin_prefetch(Ar + k + 64, 0, 1);
      v16h a = load_a_frag(Ar + k, H);
#pragma unroll
      for (int t = 0; t < 4; ++t) {
        v16h bfr = load_b_frag(Bb + (long)(n0 + 16 * t) * H + k, H);
        acc[t] = wmma16(a, bfr, acc[t]);
      }
    }
  }
  int nl = l & 15, mb = (l >> 4) * 8;
#pragma unroll
  for (int t = 0; t < 4; ++t) {
    int n = n0 + 16 * t + nl;
    float bsum = 0.f;
#pragma unroll
    for (int e = 0; e < E; ++e) bsum += outb[e * O + n];
#pragma unroll
    for (int r = 0; r < 8; ++r)
      mixed[(long)(m0 + mb + r) * O + n] = acc[t][r] + bsum;
  }
}

// ---------------------------------------------------------------------------
// Attention: one wave per (16-query tile, head, e*b). WMMA QK^T + online
// softmax + WMMA P*V (P staged through LDS to get the A-fragment layout).
// ---------------------------------------------------------------------------
__global__ __launch_bounds__(32) void k_attn(
    const _Float16* __restrict__ Qp, const _Float16* __restrict__ Kp,
    const _Float16* __restrict__ Vp, _Float16* __restrict__ ctx) {
  __shared__ __align__(32) _Float16 plds[16 * 32];
  int l = threadIdx.x & 31;
  int qt = blockIdx.x;           // 0..8 : 4 tiles stage0, 3 stage1, 2 stage2
  int head = blockIdx.y;         // 0..11
  int eb = blockIdx.z;           // 0..E*B-1
  const int qtile0[3] = {0, 4, 7}, qoff[3] = {0, 64, 112},
            tbase[3] = {0, 320, 624}, Ts[3] = {320, 304, 288};
  int stage = qt >= 7 ? 2 : (qt >= 4 ? 1 : 0);
  int qrow0 = qoff[stage] + (qt - qtile0[stage]) * 16;
  int hd0 = head * HD;
  long ebQ = (long)eb * Q * H;
  long ebT = (long)eb * KV * H;
  const _Float16* Qr = Qp + ebQ + (long)qrow0 * H + hd0;
  v16h qa0 = load_a_frag(Qr, H);
  v16h qa1 = load_a_frag(Qr + 32, H);
  for (int i = l; i < 512; i += 32) plds[i] = (_Float16)0.f;  // cols 16..31 stay 0
  float rmax[8], rsum[8];
#pragma unroll
  for (int r = 0; r < 8; ++r) { rmax[r] = -1e30f; rsum[r] = 0.f; }
  v8f cacc[4] = {};
  int T = Ts[stage], tok0 = tbase[stage];
  int mb = (l >> 4) * 8, nl = l & 15;
  for (int jt = 0; jt < T; jt += 16) {
    const _Float16* Kr = Kp + ebT + (long)(tok0 + jt) * H + hd0;
    v8f s = {};
    s = wmma16(qa0, load_b_frag(Kr, H), s);
    s = wmma16(qa1, load_b_frag(Kr + 32, H), s);
    __syncthreads();
#pragma unroll
    for (int r = 0; r < 8; ++r) {
      float sv = s[r] * 0.125f;  // 1/sqrt(HD=64)
      float bm = hred_max(sv);
      float nm = fmaxf(rmax[r], bm);
      float corr = __expf(rmax[r] - nm);
      rmax[r] = nm;
      float p = __expf(sv - nm);
      rsum[r] = rsum[r] * corr + hred_sum(p);
#pragma unroll
      for (int t = 0; t < 4; ++t) cacc[t][r] = cacc[t][r] * corr;
      plds[(mb + r) * 32 + nl] = (_Float16)p;
    }
    __syncthreads();
    v16h pa = load_a_frag(plds, 32);
#pragma unroll
    for (int t = 0; t < 4; ++t) {
      v16h vb = load_b_frag_strided(Vp + ebT, H, tok0 + jt, tok0 + T - 1,
                                    hd0 + 16 * t);
      cacc[t] = wmma16(pa, vb, cacc[t]);
    }
  }
#pragma unroll
  for (int t = 0; t < 4; ++t)
#pragma unroll
    for (int r = 0; r < 8; ++r)
      ctx[ebQ + (long)(qrow0 + mb + r) * H + hd0 + 16 * t + nl] =
          (_Float16)(cacc[t][r] / rsum[r]);
}

// ---------------------------------------------------------------------------
// Final RMSNorm over O=4096
// ---------------------------------------------------------------------------
__global__ __launch_bounds__(256) void k_rms(
    const float* __restrict__ mixed, const float* __restrict__ rw,
    const float* __restrict__ gain, float* __restrict__ out) {
  int l = threadIdx.x & 31;
  int row = blockIdx.x * 8 + (threadIdx.x >> 5);  // [0, BQ)
  const float* mr = mixed + (long)row * O;
  float s2 = 0.f;
  for (int c = l; c < O; c += 32) { float v = mr[c]; s2 += v * v; }
  s2 = wred_sum(s2);
  float rs = rsqrtf(s2 * (1.0f / O) + 1e-6f);
  for (int c = l; c < O; c += 32)
    out[(long)row * O + c] = mr[c] * rs * rw[c] * gain[c];
}

// ---------------------------------------------------------------------------
// Host launch
// ---------------------------------------------------------------------------
static inline int gsz(long n, int b) {
  long g = (n + b - 1) / b;
  return (int)(g > 262144 ? 262144 : g);
}

extern "C" void kernel_launch(void* const* d_in, const int* in_sizes, int n_in,
                              void* d_out, int out_size, void* d_ws, size_t ws_size,
                              hipStream_t stream) {
  (void)in_sizes; (void)n_in; (void)out_size; (void)ws_size;
  const float* image = (const float*)d_in[0];
  const float* gate_ln_w = (const float*)d_in[1];
  const float* gate_ln_b = (const float*)d_in[2];
  const float* gate_w = (const float*)d_in[3];
  const float* gate_b = (const float*)d_in[4];
  const float* exp_in_w = (const float*)d_in[5];
  const float* exp_in_b = (const float*)d_in[6];
  const float* exp_query = (const float*)d_in[7];
  const float* ln1_w = (const float*)d_in[8];
  const float* ln1_b = (const float*)d_in[9];
  const float* ln1kv_w = (const float*)d_in[10];
  const float* ln1kv_b = (const float*)d_in[11];
  const float* ln2_w = (const float*)d_in[12];
  const float* ln2_b = (const float*)d_in[13];
  const float* attn_in_w = (const float*)d_in[14];
  const float* attn_in_b = (const float*)d_in[15];
  const float* attn_out_w = (const float*)d_in[16];
  const float* attn_out_b = (const float*)d_in[17];
  const float* ls1 = (const float*)d_in[18];
  const float* ls2 = (const float*)d_in[19];
  const float* fc_w = (const float*)d_in[20];
  const float* fc_b = (const float*)d_in[21];
  const float* proj_w = (const float*)d_in[22];
  const float* proj_b = (const float*)d_in[23];
  const float* out_w = (const float*)d_in[24];
  const float* out_b = (const float*)d_in[25];
  const float* rms_w = (const float*)d_in[26];
  const float* output_gain = (const float*)d_in[27];
  float* outp = (float*)d_out;

  // --- workspace carve (256B aligned) ---
  char* wsb = (char*)d_ws;
  size_t off = 0;
  auto carve = [&](size_t elems, size_t esz) {
    void* p = wsb + off;
    off += ((elems * esz + 255) & ~(size_t)255);
    return p;
  };
  _Float16* exp_in_wt = (_Float16*)carve((size_t)E * H * C, 2);
  _Float16* qkv_wt    = (_Float16*)carve((size_t)E * NL * 3 * H * H, 2);
  _Float16* ao_wt     = (_Float16*)carve((size_t)E * NL * H * H, 2);
  _Float16* fc_wt     = (_Float16*)carve((size_t)E * NL * MLP * H, 2);
  _Float16* pj_wt     = (_Float16*)carve((size_t)E * NL * H * MLP, 2);
  _Float16* out_wt    = (_Float16*)carve((size_t)E * O * H, 2);
  _Float16* gated16   = (_Float16*)carve((size_t)E * BL * C, 2);
  _Float16* h16       = (_Float16*)carve((size_t)E * BL * H, 2);
  float*    t32       = (float*)   carve((size_t)E * BQ * H, 4);
  _Float16* kvn16     = (_Float16*)carve((size_t)E * B * KV * H, 2);
  _Float16* K16       = (_Float16*)carve((size_t)E * B * KV * H, 2);
  _Float16* V16       = (_Float16*)carve((size_t)E * B * KV * H + 32 * H, 2);  // +pad
  _Float16* qn16      = (_Float16*)carve((size_t)E * BQ * H, 2);
  _Float16* Qp16      = (_Float16*)carve((size_t)E * BQ * H, 2);
  _Float16* ctx16     = (_Float16*)carve((size_t)E * BQ * H, 2);
  _Float16* ln2o16    = (_Float16*)carve((size_t)E * BQ * H, 2);
  _Float16* mid16     = (_Float16*)carve((size_t)E * BQ * MLP, 2);
  _Float16* tok16     = (_Float16*)carve((size_t)E * BQ * H, 2);
  float*    mixed     = (float*)   carve((size_t)BQ * O, 4);

  // --- weight transpose + f16 conversion ---
  k_transpose16<<<gsz((long)E * C * H, 256), 256, 0, stream>>>(exp_in_w, exp_in_wt, C, H, E);
  k_transpose16<<<gsz((long)E * NL * H * 3 * H, 256), 256, 0, stream>>>(attn_in_w, qkv_wt, H, 3 * H, E * NL);
  k_transpose16<<<gsz((long)E * NL * H * H, 256), 256, 0, stream>>>(attn_out_w, ao_wt, H, H, E * NL);
  k_transpose16<<<gsz((long)E * NL * H * MLP, 256), 256, 0, stream>>>(fc_w, fc_wt, H, MLP, E * NL);
  k_transpose16<<<gsz((long)E * NL * MLP * H, 256), 256, 0, stream>>>(proj_w, pj_wt, MLP, H, E * NL);
  k_transpose16<<<gsz((long)E * H * O, 256), 256, 0, stream>>>(out_w, out_wt, H, O, E);

  // --- gating + gated activations ---
  k_gate<<<BL / 8, 256, 0, stream>>>(image, gate_ln_w, gate_ln_b, gate_w, gate_b, gated16);

  // --- h = gated @ exp_in_w + b  (per expert) ---
  k_gemm<EP_F16><<<dim3(H / 64, BL / 128, E), 128, 0, stream>>>(
      gated16, exp_in_wt, exp_in_b, h16, nullptr, nullptr, BL, H, C,
      (long)BL * C, (long)H * C, (long)H, (long)BL * H, 0);

  // --- t init = broadcast query ---
  k_bcast_query<<<gsz((long)E * BQ * H, 256), 256, 0, stream>>>(exp_query, t32, (long)E * BQ * H);

  for (int li = 0; li < NL; ++li) {
    // LayerNorms
    k_ln_kv<<<(E * B * KV) / 8, 256, 0, stream>>>(h16, exp_query, ln1kv_w, ln1kv_b, kvn16, li);
    k_ln_rows<<<(E * BQ) / 8, 256, 0, stream>>>(t32, qn16, ln1_w, ln1_b, BQ, li);
    // QKV projections
    long sBqkv = (long)NL * 3 * H * H, sBiasqkv = (long)NL * 3 * H;
    k_gemm<EP_F16><<<dim3(H / 64, BQ / 128, E), 128, 0, stream>>>(
        qn16, qkv_wt + (long)(li * 3 + 0) * H * H, attn_in_b + (long)li * 3 * H + 0 * H,
        Qp16, nullptr, nullptr, BQ, H, H, (long)BQ * H, sBqkv, sBiasqkv, (long)BQ * H, 0);
    k_gemm<EP_F16><<<dim3(H / 64, (B * KV) / 128, E), 128, 0, stream>>>(
        kvn16, qkv_wt + (long)(li * 3 + 1) * H * H, attn_in_b + (long)li * 3 * H + 1 * H,
        K16, nullptr, nullptr, B * KV, H, H, (long)B * KV * H, sBqkv, sBiasqkv, (long)B * KV * H, 0);
    k_gemm<EP_F16><<<dim3(H / 64, (B * KV) / 128, E), 128, 0, stream>>>(
        kvn16, qkv_wt + (long)(li * 3 + 2) * H * H, attn_in_b + (long)li * 3 * H + 2 * H,
        V16, nullptr, nullptr, B * KV, H, H, (long)B * KV * H, sBqkv, sBiasqkv, (long)B * KV * H, 0);
    // attention
    k_attn<<<dim3(9, NH, E * B), 32, 0, stream>>>(Qp16, K16, V16, ctx16);
    // out-proj + layer-scale residual into t32
    k_gemm<EP_RESID><<<dim3(H / 64, BQ / 128, E), 128, 0, stream>>>(
        ctx16, ao_wt + (long)li * H * H, attn_out_b + (long)li * H,
        nullptr, t32, ls1 + (long)li * H, BQ, H, H,
        (long)BQ * H, (long)NL * H * H, (long)NL * H, (long)BQ * H, (long)NL * H);
    // MLP
    k_ln_rows<<<(E * BQ) / 8, 256, 0, stream>>>(t32, ln2o16, ln2_w, ln2_b, BQ, li);
    k_gemm<EP_F16_GELU><<<dim3(MLP / 64, BQ / 128, E), 128, 0, stream>>>(
        ln2o16, fc_wt + (long)li * MLP * H, fc_b + (long)li * MLP,
        mid16, nullptr, nullptr, BQ, MLP, H,
        (long)BQ * H, (long)NL * MLP * H, (long)NL * MLP, (long)BQ * MLP, 0);
    k_gemm<EP_RESID><<<dim3(H / 64, BQ / 128, E), 128, 0, stream>>>(
        mid16, pj_wt + (long)li * H * MLP, proj_b + (long)li * H,
        nullptr, t32, ls2 + (long)li * H, BQ, H, MLP,
        (long)BQ * MLP, (long)NL * H * MLP, (long)NL * H, (long)BQ * H, (long)NL * H);
  }

  // --- output projection summed over experts + RMSNorm ---
  k_f32_to_f16<<<gsz((long)E * BQ * H, 256), 256, 0, stream>>>(t32, tok16, (long)E * BQ * H);
  k_gemm_out<<<dim3(O / 64, BQ / 64), 128, 0, stream>>>(tok16, out_wt, out_b, mixed);
  k_rms<<<BQ / 8, 256, 0, stream>>>(mixed, rms_w, output_gain, outp);
}
